// NonLocalBlock_24103356465673
// MI455X (gfx1250) — matrix operand order repaired and checked by reading, not compiled, not run on hardware
//
#include <hip/hip_runtime.h>
#include <hip/hip_bf16.h>

typedef __attribute__((ext_vector_type(16))) _Float16 v16h;
typedef __attribute__((ext_vector_type(8)))  _Float16 v8h;
typedef __attribute__((ext_vector_type(8)))  float    v8f;

#define HW   4096
#define CC   96      // attention channels
#define NN   192     // in/out channels
#define NB   8       // batch
#define JT   64      // keys staged per LDS tile
#define KP   104     // padded K row in halfs (208B -> conflict-free b128 reads)
#define VP   72      // padded V row in halfs (144B -> conflict-free b128 reads)

// A-matrix (16x32 f16) per-lane pattern: 16 contiguous halves
__device__ __forceinline__ v16h load_a16(const _Float16* __restrict__ p) {
  v8h lo = *(const v8h*)(p);
  v8h hi = *(const v8h*)(p + 8);
  return __builtin_shufflevector(lo, hi, 0,1,2,3,4,5,6,7,8,9,10,11,12,13,14,15);
}
// B-matrix (32x16 f16) per-lane pattern: 16 contiguous halves
__device__ __forceinline__ v16h load_b16(const _Float16* __restrict__ p) {
  v8h lo = *(const v8h*)(p);
  v8h hi = *(const v8h*)(p + 8);
  return __builtin_shufflevector(lo, hi, 0,1,2,3,4,5,6,7,8,9,10,11,12,13,14,15);
}

// async global -> LDS copy of 16 bytes per lane (CDNA5, tracked by ASYNCcnt)
__device__ __forceinline__ void async_b128(unsigned lds_off, const void* gptr) {
  asm volatile("global_load_async_to_lds_b128 %0, %1, off"
               :: "v"(lds_off), "v"(gptr) : "memory");
}

// ---------------- Kernel 1: out = x + pe[:, :64, :64] (residual base) -------
__global__ __launch_bounds__(256) void nlb_addpe(const float* __restrict__ x,
                                                 const float* __restrict__ pe,
                                                 float* __restrict__ out) {
  size_t idx = (size_t)blockIdx.x * 256 + threadIdx.x;       // < 8*192*4096
  int p  = (int)(idx & 4095);
  int ch = (int)((idx >> 12) % NN);
  int hh = p >> 6, ww = p & 63;
  out[idx] = x[idx] + pe[(ch * 120 + hh) * 120 + ww];
}

// ---------------- Kernel 2: QKV projection -> f16 tensors -------------------
// qt[b][i][c] (scaled by 1/64), kt[b][j][c], vv[b][c][j]
__global__ __launch_bounds__(256) void nlb_qkv(const float* __restrict__ xpe,
                                               const float* __restrict__ wq, const float* __restrict__ bq,
                                               const float* __restrict__ wk, const float* __restrict__ bk,
                                               const float* __restrict__ wv, const float* __restrict__ bv,
                                               _Float16* __restrict__ qt,
                                               _Float16* __restrict__ kt,
                                               _Float16* __restrict__ vv) {
  __shared__ float xs[NN * 64];                // 48 KB
  const int b  = blockIdx.x >> 6;
  const int p0 = (blockIdx.x & 63) << 6;
  const float* xb = xpe + (size_t)b * NN * HW + p0;
  for (int idx = threadIdx.x; idx < NN * 64; idx += 256) {
    int ch = idx >> 6, p = idx & 63;
    xs[idx] = xb[(size_t)ch * HW + p];
  }
  __syncthreads();
  for (int pair = threadIdx.x; pair < CC * 64; pair += 256) {
    int c = pair >> 6, p = pair & 63;          // c uniform per wave -> scalar weight loads
    const float* wqr = wq + c * NN;
    const float* wkr = wk + c * NN;
    const float* wvr = wv + c * NN;
    float aq = 0.f, ak = 0.f, av = 0.f;
    #pragma unroll 4
    for (int ch = 0; ch < NN; ++ch) {
      float xv = xs[(ch << 6) + p];
      aq = fmaf(wqr[ch], xv, aq);
      ak = fmaf(wkr[ch], xv, ak);
      av = fmaf(wvr[ch], xv, av);
    }
    size_t row = (size_t)b * HW + p0 + p;
    qt[row * CC + c] = (_Float16)((aq + bq[c]) * 0.015625f);   // fold 1/sqrt(HW)
    kt[row * CC + c] = (_Float16)(ak + bk[c]);
    vv[((size_t)b * CC + c) * HW + p0 + p] = (_Float16)(av + bv[c]);
  }
}

// ---------------- Kernel 3: flash attention -----------------------------------
// Block = 8 waves (one 16-row i-tile each), same batch. K/V tiles staged in LDS
// via async global->LDS DMA, double-buffered against WMMA compute.
__global__ __launch_bounds__(256) void nlb_attn(const _Float16* __restrict__ qt,
                                                const _Float16* __restrict__ kt,
                                                const _Float16* __restrict__ vv,
                                                float* __restrict__ z) {
  __shared__ _Float16 kbuf[2][JT * KP];        // 2 x 13.0 KB
  __shared__ _Float16 vbuf[2][CC * VP];        // 2 x 13.5 KB

  const int tid  = threadIdx.x;
  const int lane = tid & 31;
  const int wave = tid >> 5;
  const int b    = blockIdx.x >> 5;                       // 8 batches
  const int i0   = (((blockIdx.x & 31) << 3) + wave) << 4; // wave's 16-row i-tile
  const int hl   = lane >> 4;
  const int ln   = lane & 15;

  // issue one 64-key K/V tile into LDS buffer `bi` (6 async b128 per thread)
  auto issue_tile = [&](int jn, int bi) {
    #pragma unroll
    for (int t = 0; t < 3; ++t) {
      int chunk = tid + t * 256;                 // 768 K-chunks: row r, 12x8 halfs
      int r = chunk / 12, pr = chunk % 12;
      const _Float16* g = kt + ((size_t)b * HW + jn + r) * CC + pr * 8;
      async_b128((unsigned)(uintptr_t)&kbuf[bi][r * KP + pr * 8], g);
    }
    #pragma unroll
    for (int t = 0; t < 3; ++t) {
      int chunk = tid + t * 256;                 // 768 V-chunks: row c, 8x8 halfs
      int c = chunk >> 3, pr = chunk & 7;
      const _Float16* g = vv + ((size_t)b * CC + c) * HW + jn + pr * 8;
      async_b128((unsigned)(uintptr_t)&vbuf[bi][c * VP + pr * 8], g);
    }
  };

  // Q as B-matrix: column i = i0+ln, K-run = c in [g*32 + 16*hl, +16)
  v16h Qb[3];
  {
    const _Float16* qb = qt + ((size_t)b * HW + i0 + ln) * CC + hl * 16;
    #pragma unroll
    for (int g = 0; g < 3; ++g) Qb[g] = load_b16(qb + g * 32);
  }

  v8f acc[6];
  #pragma unroll
  for (int cg = 0; cg < 6; ++cg) acc[cg] = (v8f){};
  float m_i = -1e30f, l_i = 0.f;

  issue_tile(0, 0);
  const int steps = HW / JT;                     // 64
  for (int jb = 0; jb < steps; ++jb) {
    const int cur = jb & 1;
    asm volatile("s_wait_asynccnt 0x0" ::: "memory");  // our tile chunks landed
    __syncthreads();                                    // whole tile visible; prev compute done
    if (jb + 1 < steps) issue_tile((jb + 1) * JT, cur ^ 1);  // overlap next DMA

    #pragma unroll
    for (int sub = 0; sub < 2; ++sub) {
      const int jj = sub * 32;

      // S^T tiles: S^T[j][i] = sum_c k[j][c] * q[i][c]   (two j-subtiles)
      v8f s0 = (v8f){}, s1 = (v8f){};
      {
        const _Float16* ka0 = &kbuf[cur][(jj + ln) * KP + hl * 16];
        const _Float16* ka1 = ka0 + 16 * KP;
        #pragma unroll
        for (int g = 0; g < 3; ++g) {
          v16h K0 = load_a16(ka0 + g * 32);
          s0 = __builtin_amdgcn_wmma_f32_16x16x32_f16(false, K0, false, Qb[g],
                                                      (short)0, s0, false, false);
        }
        #pragma unroll
        for (int g = 0; g < 3; ++g) {
          v16h K1 = load_a16(ka1 + g * 32);
          s1 = __builtin_amdgcn_wmma_f32_16x16x32_f16(false, K1, false, Qb[g],
                                                      (short)0, s1, false, false);
        }
      }

      // online softmax over j for each column i (= lane%16)
      float mx = -1e30f;
      #pragma unroll
      for (int r = 0; r < 8; ++r) { mx = fmaxf(mx, s0[r]); mx = fmaxf(mx, s1[r]); }
      mx = fmaxf(mx, __shfl_xor(mx, 16, 32));
      float m_new = fmaxf(m_i, mx);
      float corr  = __expf(m_i - m_new);

      float p0[8], p1[8], sum = 0.f;
      #pragma unroll
      for (int r = 0; r < 8; ++r) {
        p0[r] = __expf(s0[r] - m_new);
        p1[r] = __expf(s1[r] - m_new);
        sum += p0[r] + p1[r];
      }
      sum += __shfl_xor(sum, 16, 32);
      l_i = l_i * corr + sum;
      m_i = m_new;
      #pragma unroll
      for (int cg = 0; cg < 6; ++cg)
        #pragma unroll
        for (int r = 0; r < 8; ++r) acc[cg][r] *= corr;

      // repack P^T (D layout) -> B-matrix layout: lane needs K = 16*hl + 0..15
      v16h Pb;
      #pragma unroll
      for (int r = 0; r < 8; ++r) {
        float x0 = __shfl_xor(p0[r], 16, 32);
        float x1 = __shfl_xor(p1[r], 16, 32);
        Pb[r]     = (_Float16)(hl ? x1 : p0[r]);   // j = 16*hl + r
        Pb[r + 8] = (_Float16)(hl ? p1[r] : x0);   // j = 16*hl + 8 + r
      }

      // O^T[c][i] += V^T-tile x P   (6 c-groups of 16 rows)
      #pragma unroll
      for (int cg = 0; cg < 6; ++cg) {
        const _Float16* va = &vbuf[cur][(cg * 16 + ln) * VP + jj + hl * 16];
        v16h Va = load_a16(va);
        acc[cg] = __builtin_amdgcn_wmma_f32_16x16x32_f16(false, Va, false, Pb,
                                                         (short)0, acc[cg], false, false);
      }
    }
  }

  float inv = 1.f / l_i;                         // replicated across lane halves
  #pragma unroll
  for (int cg = 0; cg < 6; ++cg)
    #pragma unroll
    for (int r = 0; r < 8; ++r) {
      int c = cg * 16 + r + 8 * hl;
      z[((size_t)b * CC + c) * HW + i0 + ln] = acc[cg][r] * inv;
    }
}

// ---------------- Kernel 4: out += wz @ z + bz ------------------------------
__global__ __launch_bounds__(256) void nlb_proj(const float* __restrict__ z,
                                                const float* __restrict__ wz,
                                                const float* __restrict__ bz,
                                                float* __restrict__ out) {
  __shared__ float zs[CC * 64];                // 24 KB
  const int b  = blockIdx.x >> 6;
  const int p0 = (blockIdx.x & 63) << 6;
  for (int idx = threadIdx.x; idx < CC * 64; idx += 256) {
    int c = idx >> 6, p = idx & 63;
    zs[idx] = z[((size_t)b * CC + c) * HW + p0 + p];
  }
  __syncthreads();
  for (int pair = threadIdx.x; pair < NN * 64; pair += 256) {
    int o = pair >> 6, p = pair & 63;          // o uniform per wave
    const float* wr = wz + o * CC;
    float a = bz[o];
    #pragma unroll 4
    for (int c = 0; c < CC; ++c) a = fmaf(wr[c], zs[(c << 6) + p], a);
    size_t oi = ((size_t)b * NN + o) * HW + p0 + p;
    out[oi] += a;
  }
}

extern "C" void kernel_launch(void* const* d_in, const int* in_sizes, int n_in,
                              void* d_out, int out_size, void* d_ws, size_t ws_size,
                              hipStream_t stream) {
  const float* x  = (const float*)d_in[0];
  const float* pe = (const float*)d_in[1];
  const float* wq = (const float*)d_in[2];
  const float* bq = (const float*)d_in[3];
  const float* wk = (const float*)d_in[4];
  const float* bk = (const float*)d_in[5];
  const float* wv = (const float*)d_in[6];
  const float* bv = (const float*)d_in[7];
  const float* wz = (const float*)d_in[8];
  const float* bz = (const float*)d_in[9];
  float* out = (float*)d_out;

  // workspace: z (f32) | q^T | k^T | v (f16)  -> 30 MB total
  float*    zbuf = (float*)d_ws;
  _Float16* qt = (_Float16*)((char*)d_ws + (size_t)NB * CC * HW * sizeof(float));
  _Float16* kt = qt + (size_t)NB * HW * CC;
  _Float16* vv = kt + (size_t)NB * HW * CC;

  nlb_addpe<<<(NB * NN * HW) / 256, 256, 0, stream>>>(x, pe, out);
  nlb_qkv  <<<NB * 64, 256, 0, stream>>>(out, wq, bq, wk, bk, wv, bv, qt, kt, vv);
  nlb_attn <<<NB * 32, 256, 0, stream>>>(qt, kt, vv, zbuf);
  nlb_proj <<<NB * 64, 256, 0, stream>>>(zbuf, wz, bz, out);
}
